// GCN_90752658964906
// MI455X (gfx1250) — matrix operand order repaired and checked by reading, not compiled, not run on hardware
//
#include <hip/hip_runtime.h>
#include <hip/hip_bf16.h>

typedef __attribute__((ext_vector_type(8)))  _Float16 v8h;
typedef __attribute__((ext_vector_type(16))) _Float16 v16h;
typedef __attribute__((ext_vector_type(8)))  float    v8f;

#define GCN_NODES 25
#define GCN_BATCH 1024
#define GCN_COLS  (GCN_BATCH * GCN_NODES)   // 25600
#define GCN_EPS   1e-5f

// ---------------------------------------------------------------- small prep
__global__ void k_adj_rowsum(const float* __restrict__ adj, float* __restrict__ rs) {
    int i = threadIdx.x;
    if (i < GCN_NODES) {
        float s = 0.f;
        for (int j = 0; j < GCN_NODES; ++j) s += adj[i * GCN_NODES + j];
        rs[i] = s;
    }
}

// input BatchNorm over flattened (C=3,N=25) features; output layout [col][3], f16
__global__ void k_bn_input(const float* __restrict__ x, const float* __restrict__ g,
                           const float* __restrict__ b, const float* __restrict__ m,
                           const float* __restrict__ v, _Float16* __restrict__ act) {
    int tid = blockIdx.x * blockDim.x + threadIdx.x;
    if (tid >= GCN_BATCH * 75) return;
    int bb = tid / 75, f = tid % 75;
    int c = f / GCN_NODES, n = f % GCN_NODES;
    float val = (x[tid] - m[f]) * (g[f] * rsqrtf(v[f] + GCN_EPS)) + b[f];
    act[(size_t)(bb * GCN_NODES + n) * 3 + c] = (_Float16)val;
}

// layer 0 (cin=3): fused aggregate + channel mix + bias*rowsum + relu (VALU, tiny)
__global__ void k_layer0(const _Float16* __restrict__ act, const float* __restrict__ adj,
                         const float* __restrict__ w, const float* __restrict__ bias,
                         const float* __restrict__ rowsum, _Float16* __restrict__ out) {
    int col = blockIdx.x;            // 0..25599
    int o   = threadIdx.x;           // 0..63
    int b = col / GCN_NODES, i = col % GCN_NODES;
    float h0 = 0.f, h1 = 0.f, h2 = 0.f;
    for (int j = 0; j < GCN_NODES; ++j) {
        float a = adj[i * GCN_NODES + j];
        const _Float16* hp = act + (size_t)(b * GCN_NODES + j) * 3;
        h0 += a * (float)hp[0];
        h1 += a * (float)hp[1];
        h2 += a * (float)hp[2];
    }
    float s = w[o*3+0]*h0 + w[o*3+1]*h1 + w[o*3+2]*h2 + bias[o]*rowsum[i];
    out[(size_t)col * 64 + o] = (_Float16)fmaxf(s, 0.f);
}

// node aggregation: agg[b*25+i][c] = sum_j act[b*25+j][c] * adj[i][j]
// launched with blockDim.x == C (64..1024), one block per output column
__global__ void k_aggregate(const _Float16* __restrict__ act, const float* __restrict__ adj,
                            _Float16* __restrict__ agg, int C) {
    __shared__ float arow[GCN_NODES];
    int col = blockIdx.x;
    int b = col / GCN_NODES, i = col % GCN_NODES;
    if (threadIdx.x < GCN_NODES) arow[threadIdx.x] = adj[i * GCN_NODES + threadIdx.x];
    __syncthreads();
    int c = threadIdx.x;
    const _Float16* base = act + (size_t)b * GCN_NODES * C + c;
    float s = 0.f;
    #pragma unroll
    for (int j = 0; j < GCN_NODES; ++j) s += arow[j] * (float)base[(size_t)j * C];
    agg[(size_t)col * C + c] = (_Float16)s;
}

// fp32 -> f16 weight pack
__global__ void k_pack_w(const float* __restrict__ w, _Float16* __restrict__ wp, int n) {
    int i = blockIdx.x * blockDim.x + threadIdx.x;
    if (i < n) wp[i] = (_Float16)w[i];
}

// residual conv weights with BN scale folded; rshift[o] = sc*(rb-rm)+rbeta
__global__ void k_pack_res(const float* __restrict__ rw, const float* __restrict__ rb,
                           const float* __restrict__ rg, const float* __restrict__ rbeta,
                           const float* __restrict__ rm, const float* __restrict__ rv,
                           _Float16* __restrict__ rp, float* __restrict__ rshift,
                           int cout, int cin) {
    int i = blockIdx.x * blockDim.x + threadIdx.x;
    if (i >= cout * cin) return;
    int o = i / cin, c = i % cin;
    float sc = rg[o] * rsqrtf(rv[o] + GCN_EPS);
    rp[i] = (_Float16)(rw[i] * sc);
    if (c == 0) rshift[o] = sc * (rb[o] - rm[o]) + rbeta[o];
}

// ---------------------------------------------------------------- WMMA GEMM
// Unroll-by-2 ping-pong pipeline: two fragment register sets alternate roles and
// are reloaded in place (no cross-copies, no branches in the steady-state loop);
// each WMMA group runs with the other set's loads still in flight.
//
// MODE 0 (residual pre-pass):  out[col][o] = acc + shift[o]           (no relu)
// MODE 1 (main layer):         out[col][o] = relu(acc + bias[o]*rowsum[col%25]
//                                                  + res[col*cout+o])
// CIN is compile-time: B-row stride folds into immediate load offsets.
// wave strip: 16(M) x 64(N); block = 4 waves stacked in M => 64x64 tile
template <int MODE, int CIN>
__global__ __launch_bounds__(128) void k_gemm(
    const _Float16* __restrict__ bmat,  // [col][CIN] (MODE1: aggregated, MODE0: raw act)
    const _Float16* __restrict__ wp,    // [cout][CIN] f16 weights
    const float* __restrict__ bias,     // MODE1: conv bias, MODE0: folded BN shift
    const float* __restrict__ rowsum,   // MODE1 only
    const _Float16* __restrict__ res,   // MODE1 only: residual, layout [col][cout]
    _Float16* __restrict__ out,         // [col][cout]
    int cout) {
    const int lane  = threadIdx.x & 31;
    const int wave  = threadIdx.x >> 5;
    const int tileM = blockIdx.y * 64 + wave * 16;
    const int tileN = blockIdx.x * 64;
    const int lr = lane & 15;     // A: row-in-tile, B: col-in-tile
    const int hi = lane >> 4;     // half-wave select
    const int akOff = hi << 4;    // A: +16 K for lanes 16..31
    const int bkOff = hi << 4;    // B: +16 K for lanes 16..31

    const _Float16* aBase = wp + (size_t)(tileM + lr) * CIN + akOff;
    const _Float16* bBase[4];
    #pragma unroll
    for (int t = 0; t < 4; ++t)
        bBase[t] = bmat + (size_t)(tileN + t * 16 + lr) * CIN + bkOff;

    v8f acc[4] = {};

    // two fragment sets (ping-pong, reloaded in place)
    v8h a0lo, a0hi, b0lo[4], b0hi[4];
    v8h a1lo, a1hi, b1lo[4], b1hi[4];

    auto ld0 = [&](int k) {
        a0lo = *(const v8h*)(aBase + k);
        a0hi = *(const v8h*)(aBase + k + 8);
        #pragma unroll
        for (int t = 0; t < 4; ++t) {
            b0lo[t] = *(const v8h*)(bBase[t] + k);
            b0hi[t] = *(const v8h*)(bBase[t] + k + 8);
        }
    };
    auto ld1 = [&](int k) {
        a1lo = *(const v8h*)(aBase + k);
        a1hi = *(const v8h*)(aBase + k + 8);
        #pragma unroll
        for (int t = 0; t < 4; ++t) {
            b1lo[t] = *(const v8h*)(bBase[t] + k);
            b1hi[t] = *(const v8h*)(bBase[t] + k + 8);
        }
    };
    auto mma0 = [&]() {
        union { v16h v; v8h h[2]; } af; af.h[0] = a0lo; af.h[1] = a0hi;
        #pragma unroll
        for (int t = 0; t < 4; ++t) {
            union { v16h v; v8h h[2]; } bf; bf.h[0] = b0lo[t]; bf.h[1] = b0hi[t];
            acc[t] = __builtin_amdgcn_wmma_f32_16x16x32_f16(
                false, af.v, false, bf.v, (short)0, acc[t], false, false);
        }
    };
    auto mma1 = [&]() {
        union { v16h v; v8h h[2]; } af; af.h[0] = a1lo; af.h[1] = a1hi;
        #pragma unroll
        for (int t = 0; t < 4; ++t) {
            union { v16h v; v8h h[2]; } bf; bf.h[0] = b1lo[t]; bf.h[1] = b1hi[t];
            acc[t] = __builtin_amdgcn_wmma_f32_16x16x32_f16(
                false, af.v, false, bf.v, (short)0, acc[t], false, false);
        }
    };

    constexpr int S = CIN / 64;   // all WMMA layers have CIN % 64 == 0
    ld0(0);
    #pragma unroll 1
    for (int s = 0; s < S - 1; ++s) {
        const int k0 = s * 64;
        ld1(k0 + 32);   // in flight across mma0
        mma0();
        ld0(k0 + 64);   // in flight across mma1
        mma1();
    }
    ld1(CIN - 32);
    mma0();
    mma1();

    // epilogue: D element e -> M = mbase + e, N = lr
    const int mbase = tileM + (hi << 3);
    float bia[8];
    #pragma unroll
    for (int e = 0; e < 8; ++e) bia[e] = bias[mbase + e];

    #pragma unroll
    for (int t = 0; t < 4; ++t) {
        int col = tileN + t * 16 + lr;
        _Float16 resv[8];
        if (MODE == 1) {
            float rs = rowsum[col % GCN_NODES];
            v8h idv = *(const v8h*)(res + (size_t)col * cout + mbase);
            #pragma unroll
            for (int e = 0; e < 8; ++e) {
                float xv = acc[t][e] + bia[e] * rs + (float)idv[e];
                resv[e] = (_Float16)fmaxf(xv, 0.f);
            }
        } else {
            #pragma unroll
            for (int e = 0; e < 8; ++e)
                resv[e] = (_Float16)(acc[t][e] + bia[e]);
        }
        *(v8h*)(out + (size_t)col * cout + mbase) = *(const v8h*)resv;
    }
}

// host-side dispatch over compile-time CIN
template <int MODE>
static void launch_gemm(int cin, dim3 g, hipStream_t stream,
                        const _Float16* bmat, const _Float16* wp, const float* bias,
                        const float* rowsum, const _Float16* res, _Float16* out, int cout) {
    switch (cin) {
    case 64:   k_gemm<MODE, 64>  <<<g, 128, 0, stream>>>(bmat, wp, bias, rowsum, res, out, cout); break;
    case 128:  k_gemm<MODE, 128> <<<g, 128, 0, stream>>>(bmat, wp, bias, rowsum, res, out, cout); break;
    case 256:  k_gemm<MODE, 256> <<<g, 128, 0, stream>>>(bmat, wp, bias, rowsum, res, out, cout); break;
    case 512:  k_gemm<MODE, 512> <<<g, 128, 0, stream>>>(bmat, wp, bias, rowsum, res, out, cout); break;
    case 1024: k_gemm<MODE, 1024><<<g, 128, 0, stream>>>(bmat, wp, bias, rowsum, res, out, cout); break;
    default: break;
    }
}

// ---------------------------------------------------------------- tail
__global__ void k_pool(const _Float16* __restrict__ act, float* __restrict__ pooled) {
    int b = blockIdx.x;
    for (int c = threadIdx.x; c < 1024; c += blockDim.x) {
        const _Float16* base = act + (size_t)b * GCN_NODES * 1024 + c;
        float s = 0.f;
        #pragma unroll
        for (int i = 0; i < GCN_NODES; ++i) s += (float)base[(size_t)i * 1024];
        pooled[(size_t)b * 1024 + c] = s * (1.f / GCN_NODES);
    }
}

__global__ void k_fc(const float* __restrict__ pooled, const float* __restrict__ fw,
                     const float* __restrict__ fb, float* __restrict__ out) {
    __shared__ float sp[1024];
    int b = blockIdx.x;
    for (int c = threadIdx.x; c < 1024; c += blockDim.x) sp[c] = pooled[(size_t)b * 1024 + c];
    __syncthreads();
    int k = threadIdx.x;
    if (k < 60) {
        float s = fb[k];
        for (int c = 0; c < 1024; ++c) s += sp[c] * fw[k * 1024 + c];
        out[b * 60 + k] = s;
    }
}

// ---------------------------------------------------------------- driver
extern "C" void kernel_launch(void* const* d_in, const int* in_sizes, int n_in,
                              void* d_out, int out_size, void* d_ws, size_t ws_size,
                              hipStream_t stream) {
    (void)in_sizes; (void)n_in; (void)out_size; (void)ws_size;

    static const int cfg_cin[15]  = {3,64,64,64,64,128,128,128,256,256,256,512,512,1024,1024};
    static const int cfg_cout[15] = {64,64,64,64,128,128,128,256,256,256,512,512,1024,1024,1024};
    static const int cfg_res[15]  = {0,1,1,1,2,1,1,2,1,1,2,1,2,1,1}; // 0 none, 1 id, 2 conv

    int cur = 0;
    const float* x    = (const float*)d_in[cur++];
    const float* adj  = (const float*)d_in[cur++];
    const float* bn_g = (const float*)d_in[cur++];
    const float* bn_b = (const float*)d_in[cur++];
    const float* bn_m = (const float*)d_in[cur++];
    const float* bn_v = (const float*)d_in[cur++];
    const float *lw[15], *lb[15], *lrw[15], *lrb[15], *lrg[15], *lrbeta[15], *lrm[15], *lrv[15];
    for (int l = 0; l < 15; ++l) {
        lw[l] = (const float*)d_in[cur++];
        lb[l] = (const float*)d_in[cur++];
        if (cfg_res[l] == 2) {
            lrw[l]    = (const float*)d_in[cur++];
            lrb[l]    = (const float*)d_in[cur++];
            lrg[l]    = (const float*)d_in[cur++];
            lrbeta[l] = (const float*)d_in[cur++];
            lrm[l]    = (const float*)d_in[cur++];
            lrv[l]    = (const float*)d_in[cur++];
        } else { lrw[l]=lrb[l]=lrg[l]=lrbeta[l]=lrm[l]=lrv[l]=nullptr; }
    }
    const float* fc_w = (const float*)d_in[cur++];
    const float* fc_b = (const float*)d_in[cur++];

    // workspace carve-up (offsets 256B-aligned)
    const size_t ACT = (size_t)GCN_COLS * 1024 * sizeof(_Float16);   // 50 MB
    char* w = (char*)d_ws;
    _Float16* actA   = (_Float16*)(w);
    _Float16* actB   = (_Float16*)(w + ACT);
    _Float16* agg    = (_Float16*)(w + 2 * ACT);
    _Float16* wpack  = (_Float16*)(w + 3 * ACT);
    _Float16* rpack  = (_Float16*)(w + 3 * ACT + (2u << 20));
    float*    rshift = (float*)   (w + 3 * ACT + (4u << 20));
    float*    rowsum = (float*)   (w + 3 * ACT + (4u << 20) + 4096);
    float*    pooled = (float*)   (w + 3 * ACT + (4u << 20) + 8192);

    k_adj_rowsum<<<1, 32, 0, stream>>>(adj, rowsum);
    k_bn_input<<<(GCN_BATCH * 75 + 255) / 256, 256, 0, stream>>>(x, bn_g, bn_b, bn_m, bn_v, actA);
    k_layer0<<<GCN_COLS, 64, 0, stream>>>(actA, adj, lw[0], lb[0], rowsum, actB);

    _Float16* curAct = actB;
    _Float16* nxtAct = actA;
    for (int l = 1; l < 15; ++l) {
        int cin = cfg_cin[l], cout = cfg_cout[l];
        dim3 g(GCN_COLS / 64, cout / 64);
        k_pack_w<<<(cout * cin + 255) / 256, 256, 0, stream>>>(lw[l], wpack, cout * cin);
        k_aggregate<<<GCN_COLS, cin, 0, stream>>>(curAct, adj, agg, cin);
        if (cfg_res[l] == 2) {
            // residual pre-pass R = BN(RW*h) into the output buffer; the main pass
            // reads each element and overwrites it in-place (per-thread RAW, safe)
            k_pack_res<<<(cout * cin + 255) / 256, 256, 0, stream>>>(
                lrw[l], lrb[l], lrg[l], lrbeta[l], lrm[l], lrv[l], rpack, rshift, cout, cin);
            launch_gemm<0>(cin, g, stream, curAct, rpack, rshift, nullptr, nullptr,
                           nxtAct, cout);
            launch_gemm<1>(cin, g, stream, agg, wpack, lb[l], rowsum, nxtAct,
                           nxtAct, cout);
        } else {
            // identity residual: cin == cout, stride matches
            launch_gemm<1>(cin, g, stream, agg, wpack, lb[l], rowsum, curAct,
                           nxtAct, cout);
        }
        _Float16* t = curAct; curAct = nxtAct; nxtAct = t;
    }

    k_pool<<<GCN_BATCH, 256, 0, stream>>>(curAct, pooled);
    k_fc<<<GCN_BATCH, 64, 0, stream>>>(pooled, fc_w, fc_b, (float*)d_out);
}
